// MatryoshkaDilatedCostVolume_73632919323062
// MI455X (gfx1250) — matrix operand order repaired and checked by reading, not compiled, not run on hardware
//
#include <hip/hip_runtime.h>

typedef __attribute__((ext_vector_type(16))) _Float16 v16h;
typedef __attribute__((ext_vector_type(8)))  float    v8f;
typedef __attribute__((ext_vector_type(2)))  _Float16 h2;

#define B_ 4
#define C_ 128
#define H_ 64
#define W_ 128
#define MD 4
#define SR 9
#define ND 6

struct alignas(16) Pack32 { uint4 u0, u1; };

// Row layout in LDS: x-major, 17 uint4 per x position (128 f16 channels + 8 pad
// f16 -> 272B row stride, 16B aligned so b128 LDS loads are legal).
#define CS 17

// One kernel instantiation per dilation: all offset arithmetic (o % D, o / D,
// tile bases, loop bounds) is compile-time constant.
template <int DI, int D>
__global__ __launch_bounds__(256) void matryoshka_corr_wmma(
    const float* __restrict__ x1, const float* __restrict__ x2,
    float* __restrict__ out) {
  // Tiles at column base s_j = -4D + 16j cover m+o in [-4D, 15+4D] exactly.
  constexpr int NT = (D + 1) / 2 + 1;   // ceil(D/2)+1 : {2,2,3,4,6,9}

  extern __shared__ uint4 smem[];
  uint4* ldsA = smem;              // [128][CS]  x1 row, f16
  uint4* ldsB = smem + 128 * CS;   // [256][CS]  x2 row, padded +-64, f16

  int bi = blockIdx.x;
  const int uy = bi % SR; bi /= SR;
  const int y  = bi % H_;
  const int b  = bi / H_;

  const int tid = threadIdx.x;
  const int y2  = y + (uy - MD) * D;

  float* outBase =
      out + ((size_t)((b * ND + DI) * SR + uy) * SR) * (size_t)(H_ * W_);
  float* outRow = outBase + y * W_;   // uniform; stores use 32-bit indices

  if (y2 < 0 || y2 >= H_) {
    // entire x2 row is out of bounds -> all 9*W outputs are zero (b128 stores)
    for (int j = tid; j < SR * (W_ / 4); j += 256) {
      int v = j >> 5, xi = j & 31;
      *(float4*)(outRow + (size_t)v * H_ * W_ + xi * 4) =
          float4{0.f, 0.f, 0.f, 0.f};
    }
    return;
  }

  // ---- zero the two pad column regions of ldsB (p<64 and p>=192) ----
  // Disjoint from the staged region below, so no barrier needed in between.
  for (int j = tid; j < 2 * 64 * CS; j += 256) {
    int idx = (j < 64 * CS) ? j : j + (192 - 64) * CS;
    ldsB[idx] = uint4{0u, 0u, 0u, 0u};
  }
  // ---- stage x1 row y -> ldsA (fp32 -> f16, x-major) ----
  {
    const float* p = x1 + ((size_t)(b * C_) * H_ + y) * W_;
    unsigned* l32 = (unsigned*)ldsA;
#pragma unroll 4
    for (int it = 0; it < 32; ++it) {          // 8192 packed dwords
      int idx = it * 256 + tid;
      int x  = idx & (W_ - 1);
      int cp = idx >> 7;                       // channel pair 0..63
      float g0 = p[(size_t)(2 * cp)     * H_ * W_ + x];
      float g1 = p[(size_t)(2 * cp + 1) * H_ * W_ + x];
      h2 hp = {(_Float16)g0, (_Float16)g1};
      l32[x * (CS * 4) + cp] = __builtin_bit_cast(unsigned, hp);
    }
  }
  // ---- stage x2 row y2 -> ldsB columns p=64..191 (in-bounds, no checks) ----
  {
    const float* q = x2 + ((size_t)(b * C_) * H_ + y2) * W_;
    unsigned* l32 = (unsigned*)ldsB + 64 * (CS * 4);
#pragma unroll 4
    for (int it = 0; it < 32; ++it) {          // 8192 packed dwords
      int idx = it * 256 + tid;
      int x  = idx & (W_ - 1);
      int cp = idx >> 7;
      float g0 = q[(size_t)(2 * cp)     * H_ * W_ + x];
      float g1 = q[(size_t)(2 * cp + 1) * H_ * W_ + x];
      h2 hp = {(_Float16)g0, (_Float16)g1};
      l32[x * (CS * 4) + cp] = __builtin_bit_cast(unsigned, hp);
    }
  }
  __syncthreads();

  // ---- each wave owns one 16-pixel x1 tile ----
  const int wv   = tid >> 5;   // 0..7
  const int lane = tid & 31;
  const int lo   = lane & 15;
  const int hi   = lane >> 4;
  const int x0   = wv * 16;

  // A operand (16x32 f16, M=x1 pixels, K=channels), 4 K-steps held in regs.
  // Per ISA table: lanes<16 hold K 0..7 (v0-3) & 16..23 (v4-7); lanes>=16 hold
  // K 8..15 & 24..31 for the same rows.
  v16h a[4];
  {
    const uint4* rowA = ldsA + (size_t)(x0 + lo) * CS;
#pragma unroll
    for (int ks = 0; ks < 4; ++ks) {
      uint4 a0 = rowA[ks * 4 + hi];
      uint4 a1 = rowA[ks * 4 + 2 + hi];
      a[ks] = __builtin_bit_cast(v16h, Pack32{a0, a1});
    }
  }

  // Column tiles at base s = -4D + 16j; every (m, offset o) lands in exactly
  // one tile with n = m + o + 4D - 16j in [0,16).
#pragma unroll
  for (int j = 0; j < NT; ++j) {
    constexpr int PAD = 64;
    v8f acc = {};
    // B operand (32x16 f16): lane n=lo; lanes<16 hold K 0..7 & 16..23, lanes>=16
    // hold K 8..15 & 24..31 (same order as A).
    const uint4* rowB =
        ldsB + (size_t)(x0 + lo + PAD - 4 * D + 16 * j) * CS;
#pragma unroll
    for (int ks = 0; ks < 4; ++ks) {
      uint4 b0 = rowB[ks * 4 + hi];
      uint4 b1 = rowB[ks * 4 + 2 + hi];
      v16h bv = __builtin_bit_cast(v16h, Pack32{b0, b1});
      acc = __builtin_amdgcn_wmma_f32_16x16x32_f16(
          false, a[ks], false, bv, (short)0, acc, false, false);
    }
    // Extract needed diagonals. D layout: lane holds (M = r + 8*hi, N = lo).
#pragma unroll
    for (int r = 0; r < 8; ++r) {
      const int m = r + 8 * hi;
      const int o = 16 * j - 4 * D + lo - m;   // x2 shift for this element
      if (o % D == 0) {                        // constexpr D: AND/shift/magic
        const int oq = o / D;
        if (oq >= -MD && oq <= MD) {
          float vv = acc[r];
          vv = (vv >= 0.f) ? vv : 0.1f * vv;   // leaky ReLU
          // Uniform base + lane-varying u32 index -> global_store with saddr.
          outRow[(unsigned)((oq + MD) * (H_ * W_) + x0 + m)] = vv;
        }
      }
    }
  }
}

extern "C" void kernel_launch(void* const* d_in, const int* in_sizes, int n_in,
                              void* d_out, int out_size, void* d_ws,
                              size_t ws_size, hipStream_t stream) {
  const float* x1 = (const float*)d_in[0];
  const float* x2 = (const float*)d_in[1];
  float* out = (float*)d_out;
  (void)in_sizes; (void)n_in; (void)out_size; (void)d_ws; (void)ws_size;

  const int nblocks = B_ * H_ * SR;                               // 2304
  const size_t shmem = (size_t)(128 + 256) * CS * sizeof(uint4);  // 104448 B

  matryoshka_corr_wmma<0, 1><<<nblocks, 256, shmem, stream>>>(x1, x2, out);
  matryoshka_corr_wmma<1, 2><<<nblocks, 256, shmem, stream>>>(x1, x2, out);
  matryoshka_corr_wmma<2, 3><<<nblocks, 256, shmem, stream>>>(x1, x2, out);
  matryoshka_corr_wmma<3, 5><<<nblocks, 256, shmem, stream>>>(x1, x2, out);
  matryoshka_corr_wmma<4, 9><<<nblocks, 256, shmem, stream>>>(x1, x2, out);
  matryoshka_corr_wmma<5, 16><<<nblocks, 256, shmem, stream>>>(x1, x2, out);
}
